// RelativeAttention_73693048864781
// MI455X (gfx1250) — matrix operand order repaired and checked by reading, not compiled, not run on hardware
//
#include <hip/hip_runtime.h>

#define BATCH 2
#define SEQQ 256
#define SEQK 256
#define EMB 512
#define NH 8
#define HD 64

typedef __attribute__((ext_vector_type(16))) __bf16 v16bf;
typedef __attribute__((ext_vector_type(8)))  float  v8f;

static __device__ __forceinline__ __bf16 tobf(float x) { return (__bf16)x; }

// ---- WMMA fragment builders using contiguous k = 16*(lane/16) + s (wave32) ----

// A-matrix 16x32 bf16 from row-major bf16 source
static __device__ __forceinline__ v16bf frag_a_bf16(const __bf16* base, int ld, int lane) {
  int m = lane & 15, grp = lane >> 4;
  const __bf16* row = base + (size_t)m * ld + grp * 16;
  v16bf a;
#pragma unroll
  for (int p = 0; p < 8; ++p) {
    int e = p * 2;
    a[2 * p]     = row[e];
    a[2 * p + 1] = row[e + 1];
  }
  return a;
}

// A-matrix 16x32 fragment loaded directly from row-major *f32* source with
// in-register bf16 conversion. Lane needs floats [g*16..g*16+15]
// of row m: exactly 4 contiguous float4 loads, no cross-lane exchange.
static __device__ __forceinline__ v16bf frag_a_from_f32(const float* rowbase) {
  const float4* q0 = (const float4*)rowbase;
  const float4* q1 = (const float4*)(rowbase + 8);
  float4 f0 = q0[0], f1 = q0[1], f2 = q1[0], f3 = q1[1];
  v16bf a;
  a[0]  = tobf(f0.x); a[1]  = tobf(f0.y); a[2]  = tobf(f0.z); a[3]  = tobf(f0.w);
  a[4]  = tobf(f1.x); a[5]  = tobf(f1.y); a[6]  = tobf(f1.z); a[7]  = tobf(f1.w);
  a[8]  = tobf(f2.x); a[9]  = tobf(f2.y); a[10] = tobf(f2.z); a[11] = tobf(f2.w);
  a[12] = tobf(f3.x); a[13] = tobf(f3.y); a[14] = tobf(f3.z); a[15] = tobf(f3.w);
  return a;
}

// B-matrix 32x16 bf16 from row-major B[k*ld + n]
static __device__ __forceinline__ v16bf frag_b_bf16(const __bf16* base, int ld, int lane) {
  int n = lane & 15, grp = lane >> 4;
  v16bf b;
#pragma unroll
  for (int p = 0; p < 8; ++p) {
    int k0 = grp * 16 + 2 * p;
    b[2 * p]     = base[(size_t)k0 * ld + n];
    b[2 * p + 1] = base[(size_t)(k0 + 1) * ld + n];
  }
  return b;
}

// B-matrix 32x16 where B[k,n] = W[n*ldw + k] (row-major weight, used as W^T)
static __device__ __forceinline__ v16bf frag_bT_bf16(const __bf16* base, int ldw, int lane) {
  int n = lane & 15, grp = lane >> 4;
  const __bf16* row = base + (size_t)n * ldw + grp * 16;
  v16bf b;
#pragma unroll
  for (int p = 0; p < 8; ++p) {
    b[2 * p]     = row[2 * p];
    b[2 * p + 1] = row[2 * p + 1];
  }
  return b;
}

// ---- f32 -> bf16 convert ----
__global__ __launch_bounds__(256) void cvt_f32_bf16(const float* __restrict__ src,
                                                    __bf16* __restrict__ dst, int n) {
  int i = blockIdx.x * blockDim.x + threadIdx.x;
  if (i < n) dst[i] = (__bf16)src[i];
}

// ---- generic projection GEMM: C[M,N] = A[M,K] * W[N,K]^T + bias ----
// mode 0: f32 row-major [Mtot,Ntot]
// mode 1: head-split [B,H,L,D] f32; if outBf, also bf16 of (val + rbias[c])
// mode 2: head-split transposed [B,H,D,L] f32 (for v, so p@v reads contiguous k)
__global__ __launch_bounds__(128)
void proj_gemm(const __bf16* __restrict__ A, const __bf16* __restrict__ W,
               const float* __restrict__ bias, const float* __restrict__ rbias,
               float* __restrict__ outF, __bf16* __restrict__ outBf,
               int Mtot, int Ntot, int Kd, int L, int mode) {
  int wave = threadIdx.x >> 5, lane = threadIdx.x & 31;
  int tile = blockIdx.x * (blockDim.x >> 5) + wave;
  int mtiles = Mtot >> 4;
  int tiles = mtiles * (Ntot >> 4);
  if (tile >= tiles) return;
  int mt = tile % mtiles, nt = tile / mtiles;

  v8f acc = {};
  for (int ks = 0; ks < Kd; ks += 32) {
    v16bf a = frag_a_bf16(A + (size_t)(mt * 16) * Kd + ks, Kd, lane);
    v16bf b = frag_bT_bf16(W + (size_t)(nt * 16) * Kd + ks, Kd, lane);
    acc = __builtin_amdgcn_wmma_f32_16x16x32_bf16(false, a, false, b, (short)0, acc,
                                                  false, false);
  }
  int n = lane & 15, grp = lane >> 4;
  int c = nt * 16 + n;
  float bv = bias ? bias[c] : 0.0f;
#pragma unroll
  for (int i = 0; i < 8; ++i) {
    int m = mt * 16 + i + 8 * grp;
    float v = acc[i] + bv;
    if (mode == 0) {
      outF[(size_t)m * Ntot + c] = v;
    } else {
      int bb = m / L, l = m % L;
      int h = c >> 6, d = c & 63;
      if (mode == 1) {
        size_t idx = (((size_t)(bb * NH + h)) * L + l) * HD + d;
        if (outF) outF[idx] = v;
        if (outBf) outBf[idx] = (__bf16)(v + rbias[c]);
      } else {  // mode 2: [B,H,D,L]
        outF[(((size_t)(bb * NH + h)) * HD + d) * L + l] = v;
      }
    }
  }
}

// ---- t[b,h,q,e] = sum_d qpb[b,h,q,d] * Wr[h*64+d, e]  (per (b,h): 256x64 x 64x512) ----
__global__ __launch_bounds__(128)
void t_gemm(const __bf16* __restrict__ qpb, const __bf16* __restrict__ Wr_bf,
            __bf16* __restrict__ tbuf) {
  int wave = threadIdx.x >> 5, lane = threadIdx.x & 31;
  int bh = blockIdx.y;
  int h = bh & (NH - 1);
  int tile = blockIdx.x * (blockDim.x >> 5) + wave;  // 512 tiles: mt 0..15, nt 0..31
  int mt = tile & 15, nt = tile >> 4;

  const __bf16* Abase = qpb + (size_t)bh * SEQQ * HD + (size_t)(mt * 16) * HD;
  v8f acc = {};
#pragma unroll
  for (int ks = 0; ks < 2; ++ks) {
    v16bf a = frag_a_bf16(Abase + ks * 32, HD, lane);
    v16bf b = frag_b_bf16(Wr_bf + (size_t)(h * HD + ks * 32) * EMB + nt * 16, EMB, lane);
    acc = __builtin_amdgcn_wmma_f32_16x16x32_bf16(false, a, false, b, (short)0, acc,
                                                  false, false);
  }
  int n = lane & 15, grp = lane >> 4;
#pragma unroll
  for (int i = 0; i < 8; ++i) {
    int m = mt * 16 + i + 8 * grp;
    tbuf[((size_t)bh * SEQQ + m) * EMB + nt * 16 + n] = (__bf16)acc[i];
  }
}

// ---- fused per-(b,q) attention: r*t WMMA + qk VALU + mask + softmax + p@v ----
__global__ __launch_bounds__(256)
void attn_fused(const float* __restrict__ r, const __bf16* __restrict__ tbuf,
                const float* __restrict__ qf, const float* __restrict__ kf,
                const float* __restrict__ vt, const int* __restrict__ mask,
                const float* __restrict__ rbias, const float* __restrict__ br,
                __bf16* __restrict__ xb) {
  // t slab as bf16 rows; rows 8..15 zero so B frags need no predicate.
  // B fragment for (ks, lane) is 16 *contiguous* bf16 of row n -> 2x ds_load_b128.
  __shared__ __attribute__((aligned(32))) __bf16 s_t[16 * EMB];  // 16 KB
  __shared__ float s_sc[NH][SEQK];                               // 8 KB
  __shared__ float s_qv[NH][HD];                                 // 2 KB

  int tid = threadIdx.x;
  int wave = tid >> 5, lane = tid & 31;
  int b = blockIdx.x >> 8, qi = blockIdx.x & 255;

  for (int j = tid; j < 16 * EMB; j += 256) {
    int h = j >> 9, e = j & 511;
    s_t[j] = (h < NH)
                 ? tbuf[(((size_t)(b * NH + h)) * SEQQ + qi) * EMB + e]
                 : (__bf16)0.0f;
  }
  for (int j = tid; j < NH * HD; j += 256) {
    int h = j >> 6, d = j & 63;
    s_qv[h][d] = qf[(((size_t)(b * NH + h)) * SEQQ + qi) * HD + d];
  }
  __syncthreads();

  const float* rp = r + ((size_t)(b * SEQQ + qi)) * SEQK * EMB;
  int mrow = lane & 15, grp = lane >> 4;

  // scores2^T[k,h] = r[b,q] (256x512) x t^T (512x16); 16 k-tiles over 8 waves x 2.
  // No barriers in this loop: A frags come straight from global, B frags from the
  // read-only t slab -> the 8 waves run decoupled to hide HBM latency.
  for (int tpass = 0; tpass < 2; ++tpass) {
    int kBase = (wave + tpass * NH) * 16;
    const float* arow = rp + (size_t)(kBase + mrow) * EMB + grp * 16;
    v8f acc = {};
#pragma unroll 4
    for (int ks = 0; ks < 16; ++ks) {
      const float* src = arow + ks * 32;
      __builtin_prefetch(src + 32, 0, 0);  // global_prefetch_b8 next chunk
      v16bf a   = frag_a_from_f32(src);
      v16bf bfr = *(const v16bf*)(s_t + (size_t)mrow * EMB + ks * 32 + grp * 16);
      acc = __builtin_amdgcn_wmma_f32_16x16x32_bf16(false, a, false, bfr, (short)0, acc,
                                                    false, false);
    }
    if (mrow < NH) {
#pragma unroll
      for (int i = 0; i < 8; ++i) s_sc[mrow][kBase + i + 8 * grp] = acc[i];
    }
  }
  __syncthreads();  // s_sc rows were written by multiple waves

  // qk^T (D=64) + (q+r_bias).br correction + scale + mask ; wave w owns head h=w
  const float scale = 0.125f;  // 1/sqrt(64)
  int h = wave;
  float cq = 0.0f;
  for (int d = lane; d < HD; d += 32)
    cq += (s_qv[h][d] + rbias[h * HD + d]) * br[h * HD + d];
#pragma unroll
  for (int off = 16; off > 0; off >>= 1) cq += __shfl_xor(cq, off, 32);

  for (int j = 0; j < 8; ++j) {
    int k = lane + j * 32;
    const float* kp = kf + (((size_t)(b * NH + h)) * SEQK + k) * HD;
    float qk = 0.0f;
#pragma unroll 8
    for (int d = 0; d < HD; ++d) qk += s_qv[h][d] * kp[d];
    float val = (s_sc[h][k] + qk + cq) * scale;
    if (mask[b * SEQK + k] == 1) val = -1e9f;
    s_sc[h][k] = val;
  }

  // per-wave softmax over 256 values (wave32 shfl reductions)
  float vals[8];
  float mx = -1e30f;
#pragma unroll
  for (int j = 0; j < 8; ++j) { vals[j] = s_sc[h][lane + j * 32]; mx = fmaxf(mx, vals[j]); }
#pragma unroll
  for (int off = 16; off > 0; off >>= 1) mx = fmaxf(mx, __shfl_xor(mx, off, 32));
  float sum = 0.0f;
#pragma unroll
  for (int j = 0; j < 8; ++j) {
    float e = __expf(vals[j] - mx);
    sum += e;
    s_sc[h][lane + j * 32] = e;
  }
#pragma unroll
  for (int off = 16; off > 0; off >>= 1) sum += __shfl_xor(sum, off, 32);
  float inv = 1.0f / sum;

  // x[b,h,q,d] = (p @ v); vt is [B,H,D,K] so the k-loop reads contiguous floats
#pragma unroll
  for (int dd = 0; dd < 2; ++dd) {
    int d = lane + dd * 32;
    const float* vp = vt + (((size_t)(b * NH + h)) * HD + d) * SEQK;
    float s = 0.0f;
    for (int k = 0; k < SEQK; ++k) s += s_sc[h][k] * vp[k];
    xb[((size_t)(b * SEQQ + qi)) * EMB + h * HD + d] = (__bf16)(s * inv);
  }
}

extern "C" void kernel_launch(void* const* d_in, const int* in_sizes, int n_in,
                              void* d_out, int out_size, void* d_ws, size_t ws_size,
                              hipStream_t stream) {
  (void)in_sizes; (void)n_in; (void)out_size; (void)ws_size;
  const float* query = (const float*)d_in[0];
  const float* key   = (const float*)d_in[1];
  const float* value = (const float*)d_in[2];
  const float* r     = (const float*)d_in[3];
  const int*   mask  = (const int*)d_in[4];
  const float* Wq    = (const float*)d_in[5];
  const float* bq    = (const float*)d_in[6];
  const float* Wk    = (const float*)d_in[7];
  const float* bk    = (const float*)d_in[8];
  const float* Wv    = (const float*)d_in[9];
  const float* bv    = (const float*)d_in[10];
  const float* Wr    = (const float*)d_in[11];
  const float* br    = (const float*)d_in[12];
  const float* rbias = (const float*)d_in[13];
  const float* Wo    = (const float*)d_in[14];
  const float* bo    = (const float*)d_in[15];

  char* p = (char*)d_ws;
  auto carve = [&](size_t bytes) {
    char* q = p;
    p += (bytes + 255) & ~(size_t)255;
    return q;
  };
  const int QE = BATCH * SEQQ * EMB;  // 262144
  const int EE = EMB * EMB;           // 262144

  __bf16* qx  = (__bf16*)carve((size_t)QE * 2);
  __bf16* kx  = (__bf16*)carve((size_t)QE * 2);
  __bf16* vx  = (__bf16*)carve((size_t)QE * 2);
  __bf16* wqb = (__bf16*)carve((size_t)EE * 2);
  __bf16* wkb = (__bf16*)carve((size_t)EE * 2);
  __bf16* wvb = (__bf16*)carve((size_t)EE * 2);
  __bf16* wrb = (__bf16*)carve((size_t)EE * 2);
  __bf16* wob = (__bf16*)carve((size_t)EE * 2);
  __bf16* qpb = (__bf16*)carve((size_t)QE * 2);
  float*  qfb = (float*)carve((size_t)QE * 4);
  float*  kfb = (float*)carve((size_t)QE * 4);
  float*  vtb = (float*)carve((size_t)QE * 4);
  __bf16* tb  = (__bf16*)carve((size_t)BATCH * NH * SEQQ * EMB * 2);
  __bf16* xb  = (__bf16*)carve((size_t)QE * 2);

  cvt_f32_bf16<<<QE / 256, 256, 0, stream>>>(query, qx, QE);
  cvt_f32_bf16<<<QE / 256, 256, 0, stream>>>(key, kx, QE);
  cvt_f32_bf16<<<QE / 256, 256, 0, stream>>>(value, vx, QE);
  cvt_f32_bf16<<<EE / 256, 256, 0, stream>>>(Wq, wqb, EE);
  cvt_f32_bf16<<<EE / 256, 256, 0, stream>>>(Wk, wkb, EE);
  cvt_f32_bf16<<<EE / 256, 256, 0, stream>>>(Wv, wvb, EE);
  cvt_f32_bf16<<<EE / 256, 256, 0, stream>>>(Wr, wrb, EE);
  cvt_f32_bf16<<<EE / 256, 256, 0, stream>>>(Wo, wob, EE);

  // q/k/v projections: M = B*256 = 512, N = 512, K = 512 -> 1024 tiles / 4 waves
  proj_gemm<<<256, 128, 0, stream>>>(qx, wqb, bq, rbias, qfb, qpb, 512, 512, 512, SEQQ, 1);
  proj_gemm<<<256, 128, 0, stream>>>(kx, wkb, bk, nullptr, kfb, nullptr, 512, 512, 512, SEQK, 1);
  proj_gemm<<<256, 128, 0, stream>>>(vx, wvb, bv, nullptr, vtb, nullptr, 512, 512, 512, SEQK, 2);

  // t = (q + r_bias) @ Wr_head per (b,h)
  t_gemm<<<dim3(128, BATCH * NH), 128, 0, stream>>>(qpb, wrb, tb);

  // fused scores/softmax/p@v: one WG per (b,q)
  attn_fused<<<BATCH * SEQQ, 256, 0, stream>>>(r, tb, qfb, kfb, vtb, mask, rbias, br, xb);

  // output projection to f32 d_out
  proj_gemm<<<256, 128, 0, stream>>>(xb, wob, bo, nullptr, (float*)d_out, nullptr,
                                     512, 512, 512, SEQQ, 0);
}
